// FrequencyPriorCodebookModel_51556787421598
// MI455X (gfx1250) — matrix operand order repaired and checked by reading, not compiled, not run on hardware
//
#include <hip/hip_runtime.h>
#include <math.h>

typedef __attribute__((ext_vector_type(16))) _Float16 v16h;
typedef __attribute__((ext_vector_type(8)))  float    v8f;

#define TWO_PI 6.28318530717958647692f

// ---------------------------------------------------------------------------
// FFT: row DFT (real -> half spectrum), then column DFT, ortho scaling 1/128.
// ---------------------------------------------------------------------------
__global__ void fft_rows(const float* __restrict__ x, float* __restrict__ re,
                         float* __restrict__ im, int W, int Wh) {
  const int row = blockIdx.x;                 // (b*C + c)*H + y
  __shared__ float sx[128];
  for (int i = threadIdx.x; i < W; i += blockDim.x) sx[i] = x[(long)row * W + i];
  __syncthreads();
  for (int k = threadIdx.x; k < Wh; k += blockDim.x) {
    float sre = 0.f, sim = 0.f;
    const float w0 = -TWO_PI * (float)k / (float)W;
    for (int n = 0; n < W; ++n) {
      float s, c;
      __sincosf(w0 * (float)n, &s, &c);
      sre += sx[n] * c;
      sim += sx[n] * s;
    }
    re[(long)row * Wh + k] = sre;
    im[(long)row * Wh + k] = sim;
  }
}

__global__ void fft_cols(const float* __restrict__ re1, const float* __restrict__ im1,
                         float* __restrict__ re2, float* __restrict__ im2,
                         int H, int Wh) {
  const int col = blockIdx.x;                 // (b*C + c)*Wh + k
  const int bc = col / Wh, k = col % Wh;
  __shared__ float sr[128], si[128];
  for (int y = threadIdx.x; y < H; y += blockDim.x) {
    sr[y] = re1[((long)bc * H + y) * Wh + k];
    si[y] = im1[((long)bc * H + y) * Wh + k];
  }
  __syncthreads();
  const float scale = 1.0f / 128.0f;          // ortho: 1/sqrt(H*W)
  for (int ky = threadIdx.x; ky < H; ky += blockDim.x) {
    float sre = 0.f, sim = 0.f;
    const float w0 = -TWO_PI * (float)ky / (float)H;
    for (int y = 0; y < H; ++y) {
      float s, c;
      __sincosf(w0 * (float)y, &s, &c);
      sre += sr[y] * c - si[y] * s;
      sim += sr[y] * s + si[y] * c;
    }
    re2[((long)bc * H + ky) * Wh + k] = sre * scale;
    im2[((long)bc * H + ky) * Wh + k] = sim * scale;
  }
}

// [B,3,H,Wh] complex -> [B,9,H,W] {|F|, sin(ph), cos(ph)} reflect-padded
__global__ void fft_feat(const float* __restrict__ re, const float* __restrict__ im,
                         float* __restrict__ out, int B, int C, int H, int W, int Wh,
                         long n) {
  long idx = (long)blockIdx.x * blockDim.x + threadIdx.x;
  if (idx >= n) return;
  int xo = (int)(idx % W);
  long t = idx / W;
  int y = (int)(t % H); t /= H;
  int cc = (int)(t % (3 * C));
  int b = (int)(t / (3 * C));
  int c = cc % C, typ = cc / C;
  int xi = (xo < Wh) ? xo : (W - xo);         // reflect: out[Wh+j] = in[Wh-2-j]
  long src = (((long)b * C + c) * H + y) * Wh + xi;
  float rr = re[src], ii = im[src];
  float v;
  if (typ == 0) v = sqrtf(rr * rr + ii * ii);
  else {
    float ph = atan2f(ii, rr);
    v = (typ == 1) ? sinf(ph) : cosf(ph);
  }
  out[idx] = v;
}

// ---------------------------------------------------------------------------
// Implicit-GEMM conv via V_WMMA_F32_16X16X32_F16.
//   M = Cout, N = B*Ho*Wo, K = Cin*R*S.
// Block = 128 threads = 4 waves. The 16*MT x 32 A (weight) tile is staged
// ONCE per block (all waves share tileM); each wave stages its own 32x16
// im2col B tile and computes a 16*MT x 16 output tile (MT WMMAs per k-step).
// LDS is written in fragment order so each lane reads its v16h fragment as
// two contiguous ds_load_b128. No early exit: out-of-range waves compute a
// dead tile so __syncthreads stays uniform and EXEC is all-ones at the WMMA.
// DECONV: lhs_dilation=2 gather (pad passed = 2). act: 0 none, 2 tanh.
// ---------------------------------------------------------------------------
template <int R, int S, int STRIDE, int DECONV, int MT>
__global__ __launch_bounds__(128) void conv_wmma(
    const float* __restrict__ X, const float* __restrict__ Wt,
    const float* __restrict__ Bias, float* __restrict__ Y,
    int Bn, int Cin, int Hin, int Win, int Cout,
    int pad, int Ho, int Wo, int act) {
  const int lane = threadIdx.x & 31;
  const int wave = threadIdx.x >> 5;
  __shared__ __align__(32) _Float16 ldsA[MT * 512];   // block-shared A tile
  __shared__ __align__(32) _Float16 ldsB[4][512];     // per-wave B tiles
  _Float16* lb = ldsB[wave];

  const long Npix = (long)Bn * Ho * Wo;
  const int nTilesN = (int)((Npix + 15) >> 4);
  const int tileN = blockIdx.x * 4 + wave;            // wave-uniform
  const int tileM = blockIdx.y;
  const bool valid = tileN < nTilesN;

  constexpr int RS = R * S;
  const int K = Cin * RS;
  const int Kt = (K + 31) & ~31;
  const int HW = Hin * Win;

  // ---- per-lane invariants (hoisted out of the K loop) ----
  // B staging: this lane always stages output pixel n = lane&15, and the
  // staged k values are k = 2*i + (lane>>4), i = 0..15.
  const int nn = lane & 15;
  const int khi = lane >> 4;                          // 0 or 1
  const long pix = (long)tileN * 16 + nn;
  const bool pixok = valid && (pix < Npix);
  int wo = 0, ho = 0, bidx = 0;
  if (pixok) {
    wo = (int)(pix % Wo);
    long tt = pix / Wo;
    ho = (int)(tt % Ho);
    bidx = (int)(tt / Ho);
  }
  const float* __restrict__ Xb = X + (long)bidx * Cin * HW;
  // A staging: each thread always stages k = lane, rows m = i*4 + wave.
  const int ka = lane;
  const int agrp = ka >> 4, akk = ka & 15, asub = akk >> 3;
  const int aoff = asub << 4;                         // owner-lane offset
  const int ah = (agrp << 3) + (akk & 7);             // half index within v16h
  const int mg0 = tileM * (16 * MT);

  v8f acc[MT];
#pragma unroll
  for (int mt = 0; mt < MT; ++mt) acc[mt] = v8f{0.f, 0.f, 0.f, 0.f, 0.f, 0.f, 0.f, 0.f};

  for (int k0 = 0; k0 < Kt; k0 += 32) {
    if (k0 + 32 < K)                                  // emits global_prefetch
      __builtin_prefetch(Wt + (long)mg0 * K + k0 + 32, 0, 1);

    // ---- stage A cooperatively: 16*MT rows x 32 k, fragment order ----
    const int kga = k0 + ka;
    const bool kaok = kga < K;
#pragma unroll
    for (int i = 0; i < MT * 4; ++i) {
      const int m = i * 4 + wave;                     // 0 .. 16*MT-1
      const int mg = mg0 + m;
      float v = 0.f;
      if (kaok && mg < Cout) v = Wt[(long)mg * K + kga];
      const int mt = m >> 4, mr = m & 15;
      ldsA[mt * 512 + (mr + aoff) * 16 + ah] = (_Float16)v;
    }
    // ---- stage B per wave: im2col 32 k x 16 n, k = 2*i + khi per lane ----
#pragma unroll
    for (int i = 0; i < 16; ++i) {
      const int k = 2 * i + khi;
      const int kg = k0 + k;
      float v = 0.f;
      if (pixok && kg < K) {
        const int cin = kg / RS;                      // constant division
        const int rs = kg % RS;
        const int r = rs / S, s = rs % S;
        if (!DECONV) {
          const int ih = ho * STRIDE - pad + r;
          const int iw = wo * STRIDE - pad + s;
          if (ih >= 0 && ih < Hin && iw >= 0 && iw < Win)
            v = Xb[cin * HW + ih * Win + iw];
        } else {                                      // lhs_dilation=2, pad=2
          const int ihd = ho - pad + r, iwd = wo - pad + s;
          if (ihd >= 0 && iwd >= 0 && !(ihd & 1) && !(iwd & 1)) {
            const int ih = ihd >> 1, iw = iwd >> 1;
            if (ih < Hin && iw < Win) v = Xb[cin * HW + ih * Win + iw];
          }
        }
      }
      lb[(nn + (((k >> 3) & 1) << 4)) * 16 + ((k >> 4) << 3) + (k & 7)] = (_Float16)v;
    }
    __syncthreads();                                  // A visible to all waves
    const v16h bf = *(const v16h*)(lb + lane * 16);
#pragma unroll
    for (int mt = 0; mt < MT; ++mt) {
      const v16h af = *(const v16h*)(ldsA + mt * 512 + lane * 16);
      acc[mt] = __builtin_amdgcn_wmma_f32_16x16x32_f16(
          false, af, false, bf, (short)0, acc[mt], false, false);
    }
    __syncthreads();                                  // reads done before restage
  }

  // D layout: VGPR r -> M = r + (lane<16 ? 0 : 8), N = lane & 15
  if (pixok) {
    const int mbase = (lane < 16) ? 0 : 8;
    const long outpix = ((long)bidx * Cout) * Ho * Wo + (long)ho * Wo + wo;
#pragma unroll
    for (int mt = 0; mt < MT; ++mt) {
#pragma unroll
      for (int r = 0; r < 8; ++r) {
        const int mg = mg0 + mt * 16 + mbase + r;
        if (mg < Cout) {
          float v = acc[mt][r] + Bias[mg];
          if (act == 2) v = tanhf(v);
          Y[outpix + (long)mg * Ho * Wo] = v;
        }
      }
    }
  }
}

// ---------------------------------------------------------------------------
// GroupNorm (two pass) + fused ReLU apply
// ---------------------------------------------------------------------------
__global__ __launch_bounds__(256) void gn_stats(const float* __restrict__ x,
                                                float* __restrict__ stats,
                                                int C, int HW, int groups) {
  const int bg = blockIdx.x;
  const int cpg = C / groups;
  const long n = (long)cpg * HW;
  const float* base = x + (long)bg * n;       // groups contiguous in channel dim
  float s = 0.f, s2 = 0.f;
  for (long i = threadIdx.x; i < n; i += blockDim.x) {
    float v = base[i];
    s += v; s2 += v * v;
  }
  __shared__ float rs[256], rq[256];
  rs[threadIdx.x] = s; rq[threadIdx.x] = s2;
  __syncthreads();
  for (int off = 128; off > 0; off >>= 1) {
    if ((int)threadIdx.x < off) {
      rs[threadIdx.x] += rs[threadIdx.x + off];
      rq[threadIdx.x] += rq[threadIdx.x + off];
    }
    __syncthreads();
  }
  if (threadIdx.x == 0) {
    float m = rs[0] / (float)n;
    float var = rq[0] / (float)n - m * m;
    stats[bg * 2 + 0] = m;
    stats[bg * 2 + 1] = rsqrtf(var + 1e-5f);
  }
}

__global__ void gn_apply_relu(float* __restrict__ y, const float* __restrict__ stats,
                              const float* __restrict__ gam, const float* __restrict__ bet,
                              int C, int HW, int groups, long n) {
  long idx = (long)blockIdx.x * blockDim.x + threadIdx.x;
  if (idx >= n) return;
  int c = (int)((idx / HW) % C);
  int b = (int)(idx / ((long)HW * C));
  int g = c / (C / groups);
  float m = stats[(b * groups + g) * 2 + 0];
  float rstd = stats[(b * groups + g) * 2 + 1];
  float v = (y[idx] - m) * rstd * gam[c] + bet[c];
  y[idx] = fmaxf(v, 0.f);
}

// ---------------------------------------------------------------------------
// VQ: one block per row (D=64), 512 threads = 512 codes. embed is (64,512)
// so E[j*512+t] is coalesced across t. Accumulates sum of squared diffs.
// ---------------------------------------------------------------------------
__global__ __launch_bounds__(512) void vq_kernel(const float* __restrict__ inp,
                                                 const float* __restrict__ E,
                                                 float* __restrict__ q,
                                                 float* __restrict__ lossacc) {
  const int row = blockIdx.x;
  const int t = threadIdx.x;
  __shared__ float sx[64];
  __shared__ float sd[512];
  __shared__ int si[512];
  if (t < 64) sx[t] = inp[(long)row * 64 + t];
  __syncthreads();
  float d = 0.f;
  for (int j = 0; j < 64; ++j) {
    float df = sx[j] - E[(long)j * 512 + t];
    d += df * df;
  }
  sd[t] = d; si[t] = t;
  __syncthreads();
  for (int off = 256; off > 0; off >>= 1) {
    if (t < off) {
      if (sd[t + off] < sd[t] || (sd[t + off] == sd[t] && si[t + off] < si[t])) {
        sd[t] = sd[t + off]; si[t] = si[t + off];
      }
    }
    __syncthreads();
  }
  const int best = si[0];
  __syncthreads();
  if (t < 64) {
    float e = E[(long)t * 512 + best];
    q[(long)row * 64 + t] = e;
    float df = e - sx[t];
    sd[t] = df * df;
  }
  __syncthreads();
  if (t == 0) {
    float s = 0.f;
    for (int j = 0; j < 64; ++j) s += sd[j];
    atomicAdd(lossacc, s);
  }
}

// ---------------------------------------------------------------------------
// Bilinear 2x upsample of qf (32x32 -> 64x64) + concat with qp into z[B,128,..]
// ---------------------------------------------------------------------------
__global__ void resize_concat(const float* __restrict__ qf, const float* __restrict__ qp,
                              float* __restrict__ z, long n) {
  long idx = (long)blockIdx.x * blockDim.x + threadIdx.x;
  if (idx >= n) return;
  int x = (int)(idx % 64);
  long t = idx / 64;
  int y = (int)(t % 64); t /= 64;
  int c = (int)(t % 128);
  int b = (int)(t / 128);
  float v;
  if (c < 64) {
    float fy = (y + 0.5f) * 0.5f - 0.5f;
    float fx = (x + 0.5f) * 0.5f - 0.5f;
    fy = fminf(fmaxf(fy, 0.f), 31.f);
    fx = fminf(fmaxf(fx, 0.f), 31.f);
    int y0 = (int)floorf(fy), x0 = (int)floorf(fx);
    int y1 = min(y0 + 1, 31), x1 = min(x0 + 1, 31);
    float wy = fy - y0, wx = fx - x0;
    const float* p = qf + (((long)b * 64 + c) * 32) * 32;
    float v00 = p[y0 * 32 + x0], v01 = p[y0 * 32 + x1];
    float v10 = p[y1 * 32 + x0], v11 = p[y1 * 32 + x1];
    v = v00 * (1 - wy) * (1 - wx) + v01 * (1 - wy) * wx +
        v10 * wy * (1 - wx) + v11 * wy * wx;
  } else {
    v = qp[(((long)b * 64 + (c - 64)) * 64 + y) * 64 + x];
  }
  z[idx] = v;
}

__global__ void init_loss(float* lossacc) {
  if (threadIdx.x < 2) lossacc[threadIdx.x] = 0.f;
}

__global__ void finalize_loss(const float* __restrict__ lossacc, float* __restrict__ out) {
  // vq_loss = q_loss + 0.25*e_loss; both equal mse(q, x) in forward value.
  out[0] = 1.25f * (lossacc[0] / 524288.f) + 1.25f * (lossacc[1] / 2097152.f);
}

// ---------------------------------------------------------------------------
// Host orchestration
// ---------------------------------------------------------------------------
static inline long cdivl(long a, long b) { return (a + b - 1) / b; }

#define LAUNCH_CONV(RR, SS, ST, DC, MTV, X, Wt, Bi, Y, Cin, Hin, Win, Cout, pad, Ho, Wo, act) \
  do {                                                                                        \
    long Npix_ = 8L * (Ho) * (Wo);                                                            \
    dim3 grid_((unsigned)cdivl(cdivl(Npix_, 16), 4),                                          \
               (unsigned)cdivl((Cout), 16 * (MTV)));                                          \
    conv_wmma<RR, SS, ST, DC, MTV><<<grid_, 128, 0, stream>>>(                                \
        X, Wt, Bi, Y, 8, Cin, Hin, Win, Cout, pad, Ho, Wo, act);                              \
  } while (0)

extern "C" void kernel_launch(void* const* d_in, const int* in_sizes, int n_in,
                              void* d_out, int out_size, void* d_ws, size_t ws_size,
                              hipStream_t stream) {
  (void)in_sizes; (void)n_in; (void)ws_size;
  const float* x = (const float*)d_in[0];
  // conv weights/biases (insertion order of the convs dict)
  const float* fe_c1_w = (const float*)d_in[1];  const float* fe_c1_b = (const float*)d_in[2];
  const float* fe_c2_w = (const float*)d_in[3];  const float* fe_c2_b = (const float*)d_in[4];
  const float* fe_d1_w = (const float*)d_in[5];  const float* fe_d1_b = (const float*)d_in[6];
  const float* fe_d2_w = (const float*)d_in[7];  const float* fe_d2_b = (const float*)d_in[8];
  const float* fe_p_w  = (const float*)d_in[9];  const float* fe_p_b  = (const float*)d_in[10];
  const float* pe_c1_w = (const float*)d_in[11]; const float* pe_c1_b = (const float*)d_in[12];
  const float* pe_c2_w = (const float*)d_in[13]; const float* pe_c2_b = (const float*)d_in[14];
  const float* pe_c3_w = (const float*)d_in[15]; const float* pe_c3_b = (const float*)d_in[16];
  const float* de_f_w  = (const float*)d_in[17]; const float* de_f_b  = (const float*)d_in[18];
  const float* de_c1_w = (const float*)d_in[19]; const float* de_c1_b = (const float*)d_in[20];
  const float* de_t1_w = (const float*)d_in[21]; const float* de_t1_b = (const float*)d_in[22];
  const float* de_t2_w = (const float*)d_in[23]; const float* de_t2_b = (const float*)d_in[24];
  const float* de_o_w  = (const float*)d_in[25]; const float* de_o_b  = (const float*)d_in[26];
  // group norm params
  const float* fe_g1_g = (const float*)d_in[27]; const float* fe_g1_b = (const float*)d_in[28];
  const float* fe_g2_g = (const float*)d_in[29]; const float* fe_g2_b = (const float*)d_in[30];
  const float* fe_g3_g = (const float*)d_in[31]; const float* fe_g3_b = (const float*)d_in[32];
  const float* fe_g4_g = (const float*)d_in[33]; const float* fe_g4_b = (const float*)d_in[34];
  const float* pe_g1_g = (const float*)d_in[35]; const float* pe_g1_b = (const float*)d_in[36];
  const float* pe_g2_g = (const float*)d_in[37]; const float* pe_g2_b = (const float*)d_in[38];
  const float* de_g0_g = (const float*)d_in[39]; const float* de_g0_b = (const float*)d_in[40];
  const float* de_g1_g = (const float*)d_in[41]; const float* de_g1_b = (const float*)d_in[42];
  const float* de_g2_g = (const float*)d_in[43]; const float* de_g2_b = (const float*)d_in[44];
  const float* de_g3_g = (const float*)d_in[45]; const float* de_g3_b = (const float*)d_in[46];
  const float* Ef = (const float*)d_in[47];
  const float* Ep = (const float*)d_in[48];
  float* out = (float*)d_out;

  const int B = 8;
  char* cur = (char*)d_ws;
  auto alloc = [&](long nf) {
    float* p = (float*)cur;
    cur += ((nf * 4 + 255) / 256) * 256;
    return p;
  };
  float* re1 = alloc(8L * 3 * 128 * 65);
  float* im1 = alloc(8L * 3 * 128 * 65);
  float* re2 = alloc(8L * 3 * 128 * 65);
  float* im2 = alloc(8L * 3 * 128 * 65);
  float* h9  = alloc(8L * 9 * 128 * 128);
  float* big_a = alloc(8L * 256 * 128 * 128);
  float* big_b = alloc(8L * 256 * 128 * 128);
  float* big_c = alloc(8L * 256 * 256 * 256);
  float* mid_a = alloc(8L * 256 * 64 * 64);
  float* mid_b = alloc(8L * 256 * 64 * 64);
  float* small_a = alloc(8L * 256 * 32 * 32);
  float* freq_feat = alloc(8L * 64 * 32 * 32);
  float* pix_feat  = alloc(8L * 64 * 64 * 64);
  float* qf = alloc(8L * 64 * 32 * 32);
  float* qp = alloc(8L * 64 * 64 * 64);
  float* z  = alloc(8L * 128 * 64 * 64);
  float* stats = alloc(128);
  float* lossacc = alloc(2);

  auto gnrelu = [&](float* Y, const float* g, const float* b, int HW) {
    gn_stats<<<B * 8, 256, 0, stream>>>(Y, stats, 256, HW, 8);
    long n = (long)B * 256 * HW;
    gn_apply_relu<<<(unsigned)cdivl(n, 256), 256, 0, stream>>>(Y, stats, g, b, 256, HW, 8, n);
  };

  // ---- frequency features ----
  fft_rows<<<B * 3 * 128, 128, 0, stream>>>(x, re1, im1, 128, 65);
  fft_cols<<<B * 3 * 65, 128, 0, stream>>>(re1, im1, re2, im2, 128, 65);
  {
    long n = 8L * 9 * 128 * 128;
    fft_feat<<<(unsigned)cdivl(n, 256), 256, 0, stream>>>(re2, im2, h9, B, 3, 128, 128, 65, n);
  }
  // ---- frequency encoder ----
  LAUNCH_CONV(3, 3, 1, 0, 4, h9, fe_c1_w, fe_c1_b, big_a, 9, 128, 128, 256, 1, 128, 128, 0);
  gnrelu(big_a, fe_g1_g, fe_g1_b, 128 * 128);
  LAUNCH_CONV(3, 3, 1, 0, 4, big_a, fe_c2_w, fe_c2_b, big_b, 256, 128, 128, 256, 1, 128, 128, 0);
  gnrelu(big_b, fe_g2_g, fe_g2_b, 128 * 128);
  LAUNCH_CONV(4, 4, 2, 0, 4, big_b, fe_d1_w, fe_d1_b, mid_a, 256, 128, 128, 256, 1, 64, 64, 0);
  gnrelu(mid_a, fe_g3_g, fe_g3_b, 64 * 64);
  LAUNCH_CONV(4, 4, 2, 0, 4, mid_a, fe_d2_w, fe_d2_b, small_a, 256, 64, 64, 256, 1, 32, 32, 0);
  gnrelu(small_a, fe_g4_g, fe_g4_b, 32 * 32);
  LAUNCH_CONV(1, 1, 1, 0, 4, small_a, fe_p_w, fe_p_b, freq_feat, 256, 32, 32, 64, 0, 32, 32, 0);
  // ---- pixel encoder ----
  LAUNCH_CONV(4, 4, 2, 0, 4, x, pe_c1_w, pe_c1_b, mid_b, 3, 128, 128, 256, 1, 64, 64, 0);
  gnrelu(mid_b, pe_g1_g, pe_g1_b, 64 * 64);
  LAUNCH_CONV(3, 3, 1, 0, 4, mid_b, pe_c2_w, pe_c2_b, mid_a, 256, 64, 64, 256, 1, 64, 64, 0);
  gnrelu(mid_a, pe_g2_g, pe_g2_b, 64 * 64);
  LAUNCH_CONV(3, 3, 1, 0, 4, mid_a, pe_c3_w, pe_c3_b, pix_feat, 256, 64, 64, 64, 1, 64, 64, 0);
  // ---- dual VQ ----
  init_loss<<<1, 32, 0, stream>>>(lossacc);
  vq_kernel<<<8192, 512, 0, stream>>>(freq_feat, Ef, qf, lossacc + 0);
  vq_kernel<<<32768, 512, 0, stream>>>(pix_feat, Ep, qp, lossacc + 1);
  // ---- decoder ----
  {
    long n = 8L * 128 * 64 * 64;
    resize_concat<<<(unsigned)cdivl(n, 256), 256, 0, stream>>>(qf, qp, z, n);
  }
  LAUNCH_CONV(1, 1, 1, 0, 4, z, de_f_w, de_f_b, mid_a, 128, 64, 64, 256, 0, 64, 64, 0);
  gnrelu(mid_a, de_g0_g, de_g0_b, 64 * 64);
  LAUNCH_CONV(3, 3, 1, 0, 4, mid_a, de_c1_w, de_c1_b, mid_b, 256, 64, 64, 256, 1, 64, 64, 0);
  gnrelu(mid_b, de_g1_g, de_g1_b, 64 * 64);
  LAUNCH_CONV(4, 4, 1, 1, 4, mid_b, de_t1_w, de_t1_b, big_a, 256, 64, 64, 256, 2, 128, 128, 0);
  gnrelu(big_a, de_g2_g, de_g2_b, 128 * 128);
  LAUNCH_CONV(4, 4, 1, 1, 4, big_a, de_t2_w, de_t2_b, big_c, 256, 128, 128, 256, 2, 256, 256, 0);
  gnrelu(big_c, de_g3_g, de_g3_b, 256 * 256);
  // final conv with fused tanh directly into d_out (recon = [8,3,256,256])
  LAUNCH_CONV(3, 3, 1, 0, 1, big_c, de_o_w, de_o_b, out, 256, 256, 256, 3, 1, 256, 256, 2);
  finalize_loss<<<1, 1, 0, stream>>>(lossacc, out + (out_size - 1));
}
